// OneSupremeMemory_50096498540929
// MI455X (gfx1250) — compile-verified, matrix operand order checked by reading, not run on hardware
//
#include <hip/hip_runtime.h>
#include <cstdint>
#include <cstddef>

#define B_    16
#define S_    462
#define MM_   50
#define H_    512
#define NH_   8
#define DH_   64
#define NL_   4
#define FF_   2048
#define LAT_  100
#define WB_   1000
#define VH_   500
#define NLAB_ 4
#define T_    512           // S_ + MM_
#define BT_   (B_*T_)       // 8192
#define KD_   612
#define KDP_  640
#define K2_   1224
#define K2P_  1248

typedef __bf16 bf16;
typedef __attribute__((ext_vector_type(16))) __bf16 v16bf;
typedef __attribute__((ext_vector_type(8)))  __bf16 v8bf;
typedef __attribute__((ext_vector_type(8)))  float  v8f;

// ---------------- workspace layout ----------------
static constexpr size_t AL256(size_t x){ return (x + 255) & ~size_t(255); }
constexpr size_t SZ_WQ  = (size_t)NL_*H_*H_*2;
constexpr size_t SZ_WF  = (size_t)NL_*H_*FF_*2;
constexpr size_t O_WTQ  = 0;
constexpr size_t O_WTK  = O_WTQ  + AL256(SZ_WQ);
constexpr size_t O_WTV  = O_WTK  + AL256(SZ_WQ);
constexpr size_t O_WTO  = O_WTV  + AL256(SZ_WQ);
constexpr size_t O_WTF1 = O_WTO  + AL256(SZ_WQ);
constexpr size_t O_WTF2 = O_WTF1 + AL256(SZ_WF);
constexpr size_t O_WTV1 = O_WTF2 + AL256(SZ_WF);
constexpr size_t O_WTV2 = O_WTV1 + AL256((size_t)H_*H_*2);
constexpr size_t O_WTK1 = O_WTV2 + AL256((size_t)H_*H_*2);
constexpr size_t O_WTK2 = O_WTK1 + AL256((size_t)K2_*KDP_*2);
constexpr size_t O_XF   = O_WTK2 + AL256((size_t)H_*K2P_*2);
constexpr size_t O_XB   = O_XF   + AL256((size_t)BT_*H_*4);
constexpr size_t O_QB   = O_XB   + AL256((size_t)BT_*H_*2);
constexpr size_t O_KB   = O_QB   + AL256((size_t)BT_*H_*2);
constexpr size_t O_VB   = O_KB   + AL256((size_t)BT_*H_*2);
constexpr size_t O_VT   = O_VB   + AL256((size_t)BT_*H_*2);
constexpr size_t O_SC   = O_VT   + AL256((size_t)BT_*H_*2);
constexpr size_t O_CTX  = O_SC   + AL256((size_t)B_*NH_*T_*T_*2);
constexpr size_t O_TMP  = O_CTX  + AL256((size_t)BT_*H_*2);
constexpr size_t O_FFB  = O_TMP  + AL256((size_t)BT_*H_*4);
constexpr size_t O_EXT  = O_FFB  + AL256((size_t)BT_*FF_*2);
constexpr size_t O_WTT  = O_EXT  + AL256((size_t)B_*T_*4);
constexpr size_t O_LB   = O_WTT  + AL256((size_t)B_*T_*4);
constexpr size_t O_VAL  = O_LB   + AL256((size_t)64);
constexpr size_t O_WSG  = O_VAL  + AL256((size_t)BT_*H_*4);
constexpr size_t O_CAT  = O_WSG  + AL256((size_t)BT_*H_*4);
constexpr size_t O_XC   = O_CAT  + AL256((size_t)BT_*KDP_*2);
constexpr size_t O_VH   = O_XC   + AL256((size_t)B_*3*H_*4);
constexpr size_t O_VM   = O_VH   + AL256((size_t)B_*VH_*4);
constexpr size_t O_VLV  = O_VM   + AL256((size_t)B_*LAT_*4);
constexpr size_t O_VLT  = O_VLV  + AL256((size_t)B_*LAT_*4);
constexpr size_t O_VHR  = O_VLT  + AL256((size_t)B_*LAT_*4);
constexpr size_t O_VRC  = O_VHR  + AL256((size_t)B_*VH_*4);
constexpr size_t O_T1   = O_VRC  + AL256((size_t)B_*WB_*4);
constexpr size_t O_T2   = O_T1   + AL256((size_t)B_*3*H_*4);
constexpr size_t O_T3   = O_T2   + AL256((size_t)B_*H_*4);
constexpr size_t O_ACC  = O_T3   + AL256((size_t)B_*H_*4);

// ---------------- device helpers ----------------
__device__ __forceinline__ float actf(float v, int act){
  switch(act){
    case 1: return v > 0.f ? v : 0.f;                               // relu
    case 2: return 0.5f*v*(1.f + erff(v*0.70710678118654752f));     // gelu exact
    case 3: return tanhf(v);
    case 4: return 1.f/(1.f + expf(-v));                            // sigmoid
  }
  return v;
}

// WMMA GEMM tiles
#define GBM 128
#define GBN 64
#define LDAS 40
#define LDBS 40

__device__ __forceinline__ v16bf ldfrag(const bf16* rp, int khalf){
  v8bf c0 = *(const v8bf*)(rp + khalf*8);
  v8bf c1 = *(const v8bf*)(rp + 16 + khalf*8);
  v16bf r;
#pragma unroll
  for (int i=0;i<8;++i){ r[i]=c0[i]; r[i+8]=c1[i]; }
  return r;
}

// low 32 bits of a generic pointer into LDS == wave-relative LDS byte offset (ISA 10.2)
__device__ __forceinline__ unsigned ldsaddr(const void* p){
  return (unsigned)(unsigned long long)p;
}
// CDNA5 async copy: global -> LDS, 16B per lane, tracked by ASYNCcnt
__device__ __forceinline__ void async_ld16(unsigned lds, const bf16* g){
  unsigned long long ga = (unsigned long long)g;
  asm volatile("global_load_async_to_lds_b128 %0, %1, off"
               :: "v"(lds), "v"(ga) : "memory");
}
__device__ __forceinline__ void wait_async0(){
  asm volatile("s_wait_asynccnt 0x0" ::: "memory");
}

// C = act(alpha * A(bf16,MxK) * Bt(bf16,[N][K]) + bias[n] + add[n]); batched via z = z0*nb2+z1
__global__ __launch_bounds__(256) void k_gemm(
    const bf16* __restrict__ A, long long bA, long long bA2,
    const bf16* __restrict__ Bt, long long bB, long long bB2,
    float* Cf, bf16* Cb, long long bC, long long bC2,
    const float* __restrict__ bias,
    const float* __restrict__ addv, long long bAd,
    int nb2, int lda, int ldb, int ldc,
    int Mdim, int Ndim, int Kdim, float alpha, int act)
{
  __shared__ __align__(16) bf16 SM[2*GBM*LDAS + 2*GBN*LDBS];
  bf16* As0 = SM;
  bf16* As1 = SM + GBM*LDAS;
  bf16* Bs0 = SM + 2*GBM*LDAS;
  bf16* Bs1 = SM + 2*GBM*LDAS + GBN*LDBS;

  int z = blockIdx.z;
  int z0 = z / nb2, z1 = z - z0*nb2;
  A  += (long long)z0*bA + (long long)z1*bA2;
  Bt += (long long)z0*bB + (long long)z1*bB2;
  long long coff = (long long)z0*bC + (long long)z1*bC2;
  const float* addp = addv ? (addv + (long long)z0*bAd) : nullptr;
  int m0 = blockIdx.y * GBM;
  int n0 = blockIdx.x * GBN;
  int tid  = threadIdx.x;
  int lane = tid & 31;
  int wave = tid >> 5;
  int wm = (wave >> 1) * 32;
  int wn = (wave & 1) * 32;
  int khalf = lane >> 4;
  int l15   = lane & 15;
  v8f acc[2][2];
#pragma unroll
  for (int a=0;a<2;++a)
#pragma unroll
    for (int b2=0;b2<2;++b2)
#pragma unroll
      for (int i=0;i<8;++i) acc[a][b2][i] = 0.f;

  int ksteps = Kdim >> 5;
  bool full = (m0 + GBM <= Mdim) && (n0 + GBN <= Ndim);

  if (full){
    // ---- async double-buffered path (all tiles in-bounds) ----
    int rA = tid >> 2, cA = tid & 3;           // A: this thread covers rows rA and rA+64
    const bf16* gA0 = A + (long long)(m0 + rA)*lda + cA*8;
    const bf16* gA1 = gA0 + (long long)64*lda;
    const bf16* gB  = Bt + (long long)(n0 + rA)*ldb + cA*8;   // B: 64 rows x 4 chunks
    unsigned offA = (unsigned)((rA*LDAS + cA*8)*sizeof(bf16));
    unsigned dA   = (unsigned)(64*LDAS*sizeof(bf16));
    unsigned offB = (unsigned)((rA*LDBS + cA*8)*sizeof(bf16));
    unsigned lA[2] = { ldsaddr(As0) + offA, ldsaddr(As1) + offA };
    unsigned lB[2] = { ldsaddr(Bs0) + offB, ldsaddr(Bs1) + offB };
    bf16* AsB[2] = { As0, As1 };
    bf16* BsB[2] = { Bs0, Bs1 };
    // prologue: issue K-tile 0 into buffer 0
    async_ld16(lA[0],      gA0);
    async_ld16(lA[0] + dA, gA1);
    async_ld16(lB[0],      gB);
    for (int kt = 0; kt < ksteps; ++kt){
      int buf = kt & 1;
      wait_async0();          // our wave's async copies for tile kt complete
      __syncthreads();        // everyone's copies complete -> buffer[buf] ready
      if (kt + 1 < ksteps){   // overlap next tile's copies with this tile's WMMAs
        int nb = buf ^ 1;
        int k0n = (kt + 1) << 5;
        async_ld16(lA[nb],      gA0 + k0n);
        async_ld16(lA[nb] + dA, gA1 + k0n);
        async_ld16(lB[nb],      gB  + k0n);
      }
      const bf16* Asp = AsB[buf];
      const bf16* Bsp = BsB[buf];
      v16bf aF0 = ldfrag(&Asp[(wm      + l15)*LDAS], khalf);
      v16bf aF1 = ldfrag(&Asp[(wm + 16 + l15)*LDAS], khalf);
      v16bf bF0 = ldfrag(&Bsp[(wn      + l15)*LDBS], khalf);
      v16bf bF1 = ldfrag(&Bsp[(wn + 16 + l15)*LDBS], khalf);
      acc[0][0] = __builtin_amdgcn_wmma_f32_16x16x32_bf16(false,aF0,false,bF0,(short)0,acc[0][0],false,false);
      acc[0][1] = __builtin_amdgcn_wmma_f32_16x16x32_bf16(false,aF0,false,bF1,(short)0,acc[0][1],false,false);
      acc[1][0] = __builtin_amdgcn_wmma_f32_16x16x32_bf16(false,aF1,false,bF0,(short)0,acc[1][0],false,false);
      acc[1][1] = __builtin_amdgcn_wmma_f32_16x16x32_bf16(false,aF1,false,bF1,(short)0,acc[1][1],false,false);
    }
  } else {
    // ---- guarded synchronous path (edge tiles) ----
    int brow = tid >> 2, bch = tid & 3;
    for (int kt = 0; kt < ksteps; ++kt){
      int k0 = kt << 5;
      __syncthreads();
#pragma unroll
      for (int c=0;c<2;++c){                       // A tile: 128 rows x 4 x 16B
        int idx = tid + (c<<8);
        int row = idx >> 2, ch = idx & 3;
        uint4 v = make_uint4(0u,0u,0u,0u);
        int gm = m0 + row;
        if (gm < Mdim) v = *(const uint4*)(A + (long long)gm*lda + (k0 + (ch<<3)));
        *(uint4*)&As0[row*LDAS + (ch<<3)] = v;
      }
      {                                            // B tile: 64 rows x 4 x 16B
        uint4 v = make_uint4(0u,0u,0u,0u);
        int gn = n0 + brow;
        if (gn < Ndim) v = *(const uint4*)(Bt + (long long)gn*ldb + (k0 + (bch<<3)));
        *(uint4*)&Bs0[brow*LDBS + (bch<<3)] = v;
      }
      if (kt + 1 < ksteps){
        __builtin_prefetch(A  + (long long)(m0 + brow)*lda + (k0 + 32), 0, 1);
        __builtin_prefetch(Bt + (long long)(n0 + brow)*ldb + (k0 + 32), 0, 1);
      }
      __syncthreads();
      v16bf aF0 = ldfrag(&As0[(wm      + l15)*LDAS], khalf);
      v16bf aF1 = ldfrag(&As0[(wm + 16 + l15)*LDAS], khalf);
      v16bf bF0 = ldfrag(&Bs0[(wn      + l15)*LDBS], khalf);
      v16bf bF1 = ldfrag(&Bs0[(wn + 16 + l15)*LDBS], khalf);
      acc[0][0] = __builtin_amdgcn_wmma_f32_16x16x32_bf16(false,aF0,false,bF0,(short)0,acc[0][0],false,false);
      acc[0][1] = __builtin_amdgcn_wmma_f32_16x16x32_bf16(false,aF0,false,bF1,(short)0,acc[0][1],false,false);
      acc[1][0] = __builtin_amdgcn_wmma_f32_16x16x32_bf16(false,aF1,false,bF0,(short)0,acc[1][0],false,false);
      acc[1][1] = __builtin_amdgcn_wmma_f32_16x16x32_bf16(false,aF1,false,bF1,(short)0,acc[1][1],false,false);
    }
  }

#pragma unroll
  for (int mt=0;mt<2;++mt)
#pragma unroll
    for (int nt=0;nt<2;++nt){
      int n = n0 + wn + nt*16 + l15;
      if (n >= Ndim) continue;
      float ba = bias ? bias[n] : 0.f;
      float aa = addp ? addp[n] : 0.f;
#pragma unroll
      for (int i=0;i<8;++i){
        int m = m0 + wm + mt*16 + khalf*8 + i;  // C layout: VGPR i -> M = i + 8*(lane>>4)
        if (m >= Mdim) continue;
        float v = alpha*acc[mt][nt][i] + ba + aa;
        v = actf(v, act);
        long long o = coff + (long long)m*ldc + n;
        if (Cf) Cf[o] = v;
        if (Cb) Cb[o] = (bf16)v;
      }
    }
}

// fp32 [batch][K][N] -> bf16 [batch][N][Kpad] (zero-padded K)
__global__ void k_cvtT(const float* __restrict__ in, bf16* __restrict__ o, int K, int N, int Kpad){
  long long z = blockIdx.z;
  in += z*(long long)K*N;
  o  += z*(long long)N*Kpad;
  long long i = (long long)blockIdx.x*256 + threadIdx.x;
  if (i >= (long long)N*Kpad) return;
  int k = (int)(i % Kpad);
  long long n = i / Kpad;
  o[i] = (k < K) ? (bf16)in[(long long)k*N + n] : (bf16)0.f;
}

__global__ __launch_bounds__(256) void k_lensum(const int* am, int* Lb){
  int b = blockIdx.x;
  int s = 0;
  for (int i=threadIdx.x;i<S_;i+=256) s += am[b*S_+i];
  __shared__ int red[256];
  red[threadIdx.x]=s; __syncthreads();
  for (int st=128; st>0; st>>=1){ if (threadIdx.x<st) red[threadIdx.x]+=red[threadIdx.x+st]; __syncthreads(); }
  if (threadIdx.x==0) Lb[b]=red[0];
}

// splice + embedding sum + LayerNorm -> x (f32 + bf16), ext, token-types
__global__ __launch_bounds__(256) void k_embed(const int* ids, const int* tti,
    const float* wemb, const float* pemb, const float* temb,
    const float* g, const float* bb, const float* memory, const int* Lb,
    float* xf, bf16* xb, float* ext, int* wtt)
{
  int row = blockIdx.x;
  int b = row / T_, pos = row % T_;
  int L = Lb[b];
  int wt; const float* src;
  if (pos < L){ int id = ids[b*S_+pos]; wt = tti[b*S_+pos]; src = wemb + (size_t)id*H_; }
  else if (pos < L + MM_){ wt = 1; src = memory + (size_t)(pos-L)*H_; }
  else { int s = pos - MM_; if (s > S_-1) s = S_-1; int id = ids[b*S_+s]; wt = 1; src = wemb + (size_t)id*H_; }
  int t = threadIdx.x;
  int j0 = t, j1 = t + 256;
  float v0 = src[j0] + pemb[pos*H_+j0] + temb[wt*H_+j0];
  float v1 = src[j1] + pemb[pos*H_+j1] + temb[wt*H_+j1];
  __shared__ float red[256];
  red[t] = v0+v1; __syncthreads();
  for (int s2=128; s2>0; s2>>=1){ if (t<s2) red[t]+=red[t+s2]; __syncthreads(); }
  float mean = red[0]*(1.f/H_); __syncthreads();
  float d0 = v0-mean, d1 = v1-mean;
  red[t] = d0*d0 + d1*d1; __syncthreads();
  for (int s2=128; s2>0; s2>>=1){ if (t<s2) red[t]+=red[t+s2]; __syncthreads(); }
  float is = rsqrtf(red[0]*(1.f/H_) + 1e-12f);
  float y0 = g[j0]*d0*is + bb[j0];
  float y1 = g[j1]*d1*is + bb[j1];
  size_t o = (size_t)row*H_;
  xf[o+j0]=y0; xf[o+j1]=y1;
  xb[o+j0]=(bf16)y0; xb[o+j1]=(bf16)y1;
  if (t==0){ ext[row] = (pos < L+MM_) ? 0.f : -10000.f; wtt[row] = wt; }
}

// x = LN(x + tmp)
__global__ __launch_bounds__(256) void k_addln(float* xf, bf16* xb, const float* tmp,
                                               const float* g, const float* bb){
  size_t o = (size_t)blockIdx.x*H_;
  int t = threadIdx.x;
  float v0 = xf[o+t]     + tmp[o+t];
  float v1 = xf[o+t+256] + tmp[o+t+256];
  __shared__ float red[256];
  red[t] = v0+v1; __syncthreads();
  for (int s=128; s>0; s>>=1){ if (t<s) red[t]+=red[t+s]; __syncthreads(); }
  float mean = red[0]*(1.f/H_); __syncthreads();
  float d0=v0-mean, d1=v1-mean;
  red[t] = d0*d0+d1*d1; __syncthreads();
  for (int s=128; s>0; s>>=1){ if (t<s) red[t]+=red[t+s]; __syncthreads(); }
  float is = rsqrtf(red[0]*(1.f/H_) + 1e-12f);
  float y0 = g[t]*d0*is + bb[t];
  float y1 = g[t+256]*d1*is + bb[t+256];
  xf[o+t]=y0; xf[o+t+256]=y1;
  xb[o+t]=(bf16)y0; xb[o+t+256]=(bf16)y1;
}

// v[b][t][c] -> vt[b][c][t]
__global__ void k_vtrans(const bf16* __restrict__ v, bf16* __restrict__ vt){
  long long i = (long long)blockIdx.x*256 + threadIdx.x;
  if (i >= (long long)B_*H_*T_) return;
  int tt = (int)(i % T_);
  long long r = i / T_;
  int c = (int)(r % H_);
  int b = (int)(r / H_);
  vt[i] = v[((long long)(b*T_+tt))*H_ + c];
}

// in-place row softmax on bf16 scores (row length T_)
__global__ __launch_bounds__(256) void k_softmax(bf16* sc){
  long long row = blockIdx.x;
  bf16* p = sc + row*T_;
  int t = threadIdx.x;
  float a0 = (float)p[t], a1 = (float)p[t+256];
  __shared__ float red[256];
  red[t] = fmaxf(a0,a1); __syncthreads();
  for (int s=128; s>0; s>>=1){ if (t<s) red[t]=fmaxf(red[t],red[t+s]); __syncthreads(); }
  float mx = red[0]; __syncthreads();
  float e0 = expf(a0-mx), e1 = expf(a1-mx);
  red[t] = e0+e1; __syncthreads();
  for (int s=128; s>0; s>>=1){ if (t<s) red[t]+=red[t+s]; __syncthreads(); }
  float inv = 1.f/red[0];
  p[t] = (bf16)(e0*inv); p[t+256] = (bf16)(e1*inv);
}

// cat = [x_bf16 | bf16(mean_b) | zeros]  (width KDP_)
__global__ void k_concat(const bf16* __restrict__ xb, const float* __restrict__ mean, bf16* cat){
  long long i = (long long)blockIdx.x*256 + threadIdx.x;
  if (i >= (long long)BT_*KDP_) return;
  int j = (int)(i % KDP_);
  long long bt = i / KDP_;
  int b = (int)(bt / T_);
  bf16 v;
  if (j < H_)       v = xb[bt*H_ + j];
  else if (j < KD_) v = (bf16)mean[b*LAT_ + (j-H_)];
  else              v = (bf16)0.f;
  cat[i] = v;
}

// zero padded columns [K2_..K2P_) of the [BT_][K2P_] view
__global__ void k_zerotail(bf16* p){
  long long i = (long long)blockIdx.x*256 + threadIdx.x;
  const int pad = K2P_ - K2_;
  if (i >= (long long)BT_*pad) return;
  long long r = i / pad;
  int c = K2_ + (int)(i % pad);
  p[r*K2P_ + c] = (bf16)0.f;
}

// fw = softmax over tokens (masked where wt==1) ; out = relu(sum_t fw*value)
__global__ __launch_bounds__(256) void k_fwval(const float* __restrict__ w, const float* __restrict__ val,
                                               const int* __restrict__ wtt, float* xc, int col){
  int b = blockIdx.x >> 1;
  int n = (blockIdx.x & 1)*256 + threadIdx.x;
  float mx = -3.0e38f;
  for (int t=0;t<T_;++t)
    if (wtt[b*T_+t] != 1) mx = fmaxf(mx, w[((long long)(b*T_+t))*H_ + n]);
  float se=0.f, sv=0.f;
  for (int t=0;t<T_;++t){
    if (wtt[b*T_+t] == 1) continue;
    float e = expf(w[((long long)(b*T_+t))*H_ + n] - mx);
    se += e;
    sv += e * val[((long long)(b*T_+t))*H_ + n];
  }
  float r = sv/se;
  xc[b*(3*H_) + col + n] = r > 0.f ? r : 0.f;
}

// small fp32 GEMM: block-per-output reduction
__global__ __launch_bounds__(128) void k_sgemm(const float* __restrict__ A, int lda,
                                               const float* __restrict__ W, int ldw,
                                               const float* __restrict__ bias,
                                               float* out, int ldo, int Ni, int Ki, int act){
  int o = blockIdx.x;
  int m = o / Ni, n = o - m*Ni;
  float s = 0.f;
  for (int k=threadIdx.x; k<Ki; k+=128)
    s += A[(long long)m*lda + k] * W[(long long)k*ldw + n];
  __shared__ float red[128];
  red[threadIdx.x]=s; __syncthreads();
  for (int st=64; st>0; st>>=1){ if (threadIdx.x<st) red[threadIdx.x]+=red[threadIdx.x+st]; __syncthreads(); }
  if (threadIdx.x==0){
    float v = red[0] + (bias ? bias[n] : 0.f);
    out[(long long)m*ldo + n] = actf(v, act);
  }
}

__global__ void k_latent(const float* mean, const float* lv, const float* eps, float* lat){
  int i = blockIdx.x*256 + threadIdx.x;
  if (i >= B_*LAT_) return;
  lat[i] = mean[i] + eps[i]*expf(0.5f*lv[i]);
}

// batchnorm over batch (population var), in-place
__global__ void k_bn(float* x, const float* g, const float* bb, int N){
  int n = blockIdx.x*256 + threadIdx.x;
  if (n >= N) return;
  float m=0.f;
  for (int r=0;r<B_;++r) m += x[r*N+n];
  m *= (1.f/B_);
  float v=0.f;
  for (int r=0;r<B_;++r){ float d = x[r*N+n]-m; v += d*d; }
  v *= (1.f/B_);
  float is = rsqrtf(v + 1e-5f);
  for (int r=0;r<B_;++r) x[r*N+n] = g[n]*(x[r*N+n]-m)*is + bb[n];
}

__global__ __launch_bounds__(256) void k_lossA(const float* rc, const float* wb, float* acc){
  float s=0.f;
  for (int i=threadIdx.x; i<B_*WB_; i+=256){ float d = rc[i]-wb[i]; s += d*d; }
  __shared__ float red[256];
  red[threadIdx.x]=s; __syncthreads();
  for (int st=128; st>0; st>>=1){ if (threadIdx.x<st) red[threadIdx.x]+=red[threadIdx.x+st]; __syncthreads(); }
  if (threadIdx.x==0) acc[0] = red[0]*(1.f/(B_*WB_));
}

__global__ __launch_bounds__(256) void k_lossB(const float* mean, const float* lv, float* acc){
  float s=0.f;
  for (int i=threadIdx.x; i<B_*LAT_; i+=256)
    s += 1.f + lv[i] - mean[i]*mean[i] - expf(lv[i]);
  __shared__ float red[256];
  red[threadIdx.x]=s; __syncthreads();
  for (int st=128; st>0; st>>=1){ if (threadIdx.x<st) red[threadIdx.x]+=red[threadIdx.x+st]; __syncthreads(); }
  if (threadIdx.x==0) acc[1] = red[0];
}

__global__ void k_final(const float* acc, float* out){
  out[16*NLAB_] = acc[0] - 0.5f*acc[1];
}

// ---------------- host ----------------
extern "C" void kernel_launch(void* const* d_in, const int* in_sizes, int n_in,
                              void* d_out, int out_size, void* d_ws, size_t ws_size,
                              hipStream_t stream)
{
  (void)in_sizes; (void)n_in; (void)out_size; (void)ws_size;
  auto f  = [&](int i){ return (const float*)d_in[i]; };
  auto ip = [&](int i){ return (const int*)d_in[i]; };
  const int *qid=ip(0), *qtt=ip(1), *qam=ip(2);
  const int *aid=ip(3), *att=ip(4), *aam=ip(5);
  const int *bid=ip(6), *btt=ip(7), *bam=ip(8);
  const float *wordbag=f(9), *vaeeps=f(10);
  const float *wemb=f(11), *pemb=f(12), *temb=f(13), *elng=f(14), *elnb=f(15);
  const float *Wq=f(16), *Wk=f(17), *Wv=f(18), *Wo=f(19), *Wf1=f(20), *Wf2=f(21);
  const float *bq=f(22), *bk=f(23), *bv=f(24), *bo=f(25), *bf1=f(26), *bf2=f(27);
  const float *ln1b=f(28), *ln2b=f(29), *ln1g=f(30), *ln2g=f(31);
  const float *memq=f(32), *mema=f(33);
  const float *vew1=f(34), *vemw=f(35), *vevw=f(36), *vdw1=f(37), *vdw2=f(38);
  const float *kw1=f(39), *kw2=f(40), *vw1=f(41), *vw2=f(42);
  const float *cw1=f(43), *cw2=f(44), *cw3=f(45), *cw4=f(46);
  const float *veb1=f(47), *vemb=f(48), *vevb=f(49), *vdb1=f(50), *vdb2=f(51);
  const float *kb1=f(52), *kb2=f(53), *vb1=f(54), *vb2=f(55);
  const float *cb1=f(56), *cb2=f(57), *cb3=f(58), *cb4=f(59);
  const float *bn1b=f(60), *bn2b=f(61), *bn3b=f(62);
  const float *bn1g=f(63), *bn2g=f(64), *bn3g=f(65);

  char* ws = (char*)d_ws;
  bf16 *WTQ=(bf16*)(ws+O_WTQ), *WTK=(bf16*)(ws+O_WTK), *WTV=(bf16*)(ws+O_WTV), *WTO=(bf16*)(ws+O_WTO);
  bf16 *WTF1=(bf16*)(ws+O_WTF1), *WTF2=(bf16*)(ws+O_WTF2);
  bf16 *WTV1=(bf16*)(ws+O_WTV1), *WTV2=(bf16*)(ws+O_WTV2), *WTK1=(bf16*)(ws+O_WTK1), *WTK2=(bf16*)(ws+O_WTK2);
  float *XF=(float*)(ws+O_XF);
  bf16 *XB=(bf16*)(ws+O_XB), *QB=(bf16*)(ws+O_QB), *KB=(bf16*)(ws+O_KB), *VBUF=(bf16*)(ws+O_VB);
  bf16 *VT=(bf16*)(ws+O_VT), *SC=(bf16*)(ws+O_SC), *CTX=(bf16*)(ws+O_CTX);
  float *TMP=(float*)(ws+O_TMP);
  bf16 *FFB=(bf16*)(ws+O_FFB);
  float *EXT=(float*)(ws+O_EXT);
  int *WTT=(int*)(ws+O_WTT), *LB=(int*)(ws+O_LB);
  float *VALB=(float*)(ws+O_VAL), *WSG=(float*)(ws+O_WSG);
  bf16 *CAT=(bf16*)(ws+O_CAT);
  float *XC=(float*)(ws+O_XC);
  float *VAEH=(float*)(ws+O_VH), *VMEAN=(float*)(ws+O_VM), *VLV=(float*)(ws+O_VLV);
  float *VLT=(float*)(ws+O_VLT), *VHR=(float*)(ws+O_VHR), *VRC=(float*)(ws+O_VRC);
  float *T1B=(float*)(ws+O_T1), *T2B=(float*)(ws+O_T2), *T3B=(float*)(ws+O_T3);
  float *ACC=(float*)(ws+O_ACC);
  float *out = (float*)d_out;

  // ---- weight convert + transpose to bf16 [N][Kpad] ----
  auto cvt = [&](const float* in, bf16* o, int K, int N, int Kpad, int batch){
    long long tot = (long long)N*Kpad;
    dim3 g((unsigned)((tot+255)/256), 1, (unsigned)batch);
    k_cvtT<<<g,256,0,stream>>>(in,o,K,N,Kpad);
  };
  cvt(Wq,  WTQ,  H_, H_,  H_,  NL_);
  cvt(Wk,  WTK,  H_, H_,  H_,  NL_);
  cvt(Wv,  WTV,  H_, H_,  H_,  NL_);
  cvt(Wo,  WTO,  H_, H_,  H_,  NL_);
  cvt(Wf1, WTF1, H_, FF_, H_,  NL_);
  cvt(Wf2, WTF2, FF_,H_,  FF_, NL_);
  cvt(vw1, WTV1, H_, H_,  H_,  1);
  cvt(vw2, WTV2, H_, H_,  H_,  1);
  cvt(kw1, WTK1, KD_,K2_, KDP_,1);
  cvt(kw2, WTK2, K2_,H_,  K2P_,1);

  // ---- VAE ----
  auto sg = [&](const float* A,int lda,const float* W,int ldw,const float* bias,
                float* o,int ldo,int M,int N,int K,int act){
    k_sgemm<<<dim3((unsigned)(M*N)),128,0,stream>>>(A,lda,W,ldw,bias,o,ldo,N,K,act);
  };
  sg(wordbag,WB_,  vew1,VH_,  veb1, VAEH, VH_,  B_, VH_,  WB_, 1);
  sg(VAEH,   VH_,  vemw,LAT_, vemb, VMEAN,LAT_, B_, LAT_, VH_, 0);
  sg(VAEH,   VH_,  vevw,LAT_, vevb, VLV,  LAT_, B_, LAT_, VH_, 0);
  k_latent<<<dim3((B_*LAT_+255)/256),256,0,stream>>>(VMEAN,VLV,vaeeps,VLT);
  sg(VLT,    LAT_, vdw1,VH_,  vdb1, VHR,  VH_,  B_, VH_,  LAT_,1);
  sg(VHR,    VH_,  vdw2,WB_,  vdb2, VRC,  WB_,  B_, WB_,  VH_, 4);
  k_lossA<<<1,256,0,stream>>>(VRC,wordbag,ACC);
  k_lossB<<<1,256,0,stream>>>(VMEAN,VLV,ACC);

  // ---- WMMA GEMM launcher ----
  auto gemm = [&](const bf16* A,long long bA,long long bA2,
                  const bf16* Bm,long long bB,long long bB2,
                  float* Cf,bf16* Cb,long long bC,long long bC2,
                  const float* bias,const float* addv,long long bAd,
                  int nb2,int lda,int ldb,int ldc,int M,int N,int K,
                  float al,int act,int batch){
    dim3 g((unsigned)((N+GBN-1)/GBN),(unsigned)((M+GBM-1)/GBM),(unsigned)batch);
    k_gemm<<<g,256,0,stream>>>(A,bA,bA2,Bm,bB,bB2,Cf,Cb,bC,bC2,bias,addv,bAd,
                               nb2,lda,ldb,ldc,M,N,K,al,act);
  };

  // ---- one passage encoder + gated pooling ----
  auto rep = [&](const int* ids,const int* tti,const int* am,const float* memory,int col){
    k_lensum<<<B_,256,0,stream>>>(am,LB);
    k_embed<<<B_*T_,256,0,stream>>>(ids,tti,wemb,pemb,temb,elng,elnb,memory,LB,XF,XB,EXT,WTT);
    for (int l=0;l<NL_;++l){
      const bf16 *wq  = WTQ  + (size_t)l*H_*H_;
      const bf16 *wk  = WTK  + (size_t)l*H_*H_;
      const bf16 *wv  = WTV  + (size_t)l*H_*H_;
      const bf16 *wo  = WTO  + (size_t)l*H_*H_;
      const bf16 *wf1 = WTF1 + (size_t)l*FF_*H_;
      const bf16 *wf2 = WTF2 + (size_t)l*H_*FF_;
      gemm(XB,0,0, wq,0,0, nullptr,QB,  0,0, bq+l*H_, nullptr,0, 1, H_,H_,H_, BT_,H_,H_, 1.f,0, 1);
      gemm(XB,0,0, wk,0,0, nullptr,KB,  0,0, bk+l*H_, nullptr,0, 1, H_,H_,H_, BT_,H_,H_, 1.f,0, 1);
      gemm(XB,0,0, wv,0,0, nullptr,VBUF,0,0, bv+l*H_, nullptr,0, 1, H_,H_,H_, BT_,H_,H_, 1.f,0, 1);
      k_vtrans<<<dim3((unsigned)(((long long)B_*H_*T_+255)/256)),256,0,stream>>>(VBUF,VT);
      // scores[z] = 0.125 * Q Kt + ext   (z = b*NH + h)
      gemm(QB,(long long)T_*H_,DH_,  KB,(long long)T_*H_,DH_,
           nullptr,SC,(long long)NH_*T_*T_,(long long)T_*T_,
           nullptr, EXT,(long long)T_,
           NH_, H_,H_,T_, T_,T_,DH_, 0.125f,0, B_*NH_);
      k_softmax<<<B_*NH_*T_,256,0,stream>>>(SC);
      // ctx[z] = attn @ V (via transposed V)
      gemm(SC,(long long)NH_*T_*T_,(long long)T_*T_,  VT,(long long)H_*T_,(long long)DH_*T_,
           nullptr,CTX,(long long)T_*H_,DH_,
           nullptr, nullptr,0,
           NH_, T_,T_,H_, T_,DH_,T_, 1.f,0, B_*NH_);
      gemm(CTX,0,0, wo,0,0, TMP,nullptr,0,0, bo+l*H_, nullptr,0, 1, H_,H_,H_, BT_,H_,H_, 1.f,0, 1);
      k_addln<<<BT_,256,0,stream>>>(XF,XB,TMP,ln1g+l*H_,ln1b+l*H_);
      gemm(XB,0,0,  wf1,0,0, nullptr,FFB,0,0, bf1+l*FF_, nullptr,0, 1, H_, H_, FF_, BT_,FF_,H_, 1.f,2, 1);
      gemm(FFB,0,0, wf2,0,0, TMP,nullptr,0,0, bf2+l*H_,  nullptr,0, 1, FF_,FF_,H_,  BT_,H_,FF_, 1.f,0, 1);
      k_addln<<<BT_,256,0,stream>>>(XF,XB,TMP,ln2g+l*H_,ln2b+l*H_);
    }
    // value = tanh(relu(x@v_w1+b)@v_w2+b)
    gemm(XB,0,0, WTV1,0,0, nullptr,QB,0,0, vb1, nullptr,0, 1, H_,H_,H_, BT_,H_,H_, 1.f,1, 1);
    gemm(QB,0,0, WTV2,0,0, VALB,nullptr,0,0, vb2, nullptr,0, 1, H_,H_,H_, BT_,H_,H_, 1.f,3, 1);
    // w = sigmoid(relu([x|mean]@k_w1+b)@k_w2+b)
    k_concat<<<dim3((unsigned)(((long long)BT_*KDP_+255)/256)),256,0,stream>>>(XB,VMEAN,CAT);
    k_zerotail<<<dim3((unsigned)(((long long)BT_*(K2P_-K2_)+255)/256)),256,0,stream>>>(FFB);
    gemm(CAT,0,0, WTK1,0,0, nullptr,FFB,0,0, kb1, nullptr,0, 1, KDP_,KDP_,K2P_, BT_,K2_,KDP_, 1.f,1, 1);
    gemm(FFB,0,0, WTK2,0,0, WSG,nullptr,0,0, kb2, nullptr,0, 1, K2P_,K2P_,H_,  BT_,H_,K2P_, 1.f,4, 1);
    k_fwval<<<B_*2,256,0,stream>>>(WSG,VALB,WTT,XC,col);
  };
  rep(qid,qtt,qam,memq,0);
  rep(aid,att,aam,mema,H_);
  rep(bid,btt,bam,memq,2*H_);

  // ---- classifier ----
  sg(XC, 3*H_, cw1,3*H_, cb1, T1B,3*H_, B_,3*H_,3*H_, 1);
  k_bn<<<dim3((3*H_+255)/256),256,0,stream>>>(T1B,bn1g,bn1b,3*H_);
  sg(T1B,3*H_, cw2,H_,   cb2, T2B,H_,   B_,H_,  3*H_, 1);
  k_bn<<<dim3((H_+255)/256),256,0,stream>>>(T2B,bn2g,bn2b,H_);
  sg(T2B,H_,   cw3,H_,   cb3, T3B,H_,   B_,H_,  H_,   1);
  k_bn<<<dim3((H_+255)/256),256,0,stream>>>(T3B,bn3g,bn3b,H_);
  sg(T3B,H_,   cw4,NLAB_,cb4, out,NLAB_,B_,NLAB_,H_,  0);
  k_final<<<1,1,0,stream>>>(ACC,out);
}
